// ESCL_54846732370342
// MI455X (gfx1250) — compile-verified, matrix operand order checked by reading, not run on hardware
//
#include <hip/hip_runtime.h>
#include <hip/hip_bf16.h>
#include <math.h>

// Supervised-contrastive-style loss:
//   f = normalize(features); S = f @ f^T; sim = exp(S/0.07)
//   pos_i = sum_{j!=i} sim_ij * (eq_ij + 1e-7); tot_i = sum_{j!=i} sim_ij
//   loss = -mean(log(pos/tot))   (note pos+neg == tot)
//
// Strategy: fp32 -> (hi,lo) f16 split, S ~= hi*hi + hi*lo + lo*hi via
// v_wmma_f32_16x16x32_f16 (3 WMMAs per 16x16x32 chunk, near-fp32 accuracy).
// Compute-bound: 68.7 GFLOP, inputs fit in L2 (16 MB << 192 MB).

typedef __attribute__((ext_vector_type(16))) _Float16 v16h;
typedef __attribute__((ext_vector_type(8)))  _Float16 v8h;
typedef __attribute__((ext_vector_type(8)))  float    v8f;

#define NROWS 8192
#define DIMS  512
#define KCHUNKS (DIMS / 32)        // 16 K-chunks of 32
#define JCHUNKS 8                  // waves per block; each wave owns 1024 cols
#define STRIPS_PER_CHUNK 16        // 16 strips * 64 cols = 1024 cols
#define INV_T (1.0f / 0.07f)
#define MASK_EPS 1e-7f

// ---------------- kernel 1: row L2-normalize + f16 hi/lo split ------------
__global__ void __launch_bounds__(256)
ESCL_norm_kernel(const float* __restrict__ feat,
                 _Float16* __restrict__ hi, _Float16* __restrict__ lo) {
  __shared__ float red[256];
  const int row = blockIdx.x;
  const int tid = threadIdx.x;
  const float* fr = feat + (size_t)row * DIMS;
  const float a0 = fr[tid];
  const float a1 = fr[tid + 256];
  red[tid] = a0 * a0 + a1 * a1;
  __syncthreads();
  for (int s = 128; s > 0; s >>= 1) {
    if (tid < s) red[tid] += red[tid + s];
    __syncthreads();
  }
  const float inv = 1.0f / fmaxf(sqrtf(red[0]), 1e-12f);
  const float n0 = a0 * inv, n1 = a1 * inv;
  const _Float16 h0 = (_Float16)n0;
  const _Float16 h1 = (_Float16)n1;
  const size_t base = (size_t)row * DIMS;
  hi[base + tid]       = h0;
  hi[base + tid + 256] = h1;
  lo[base + tid]       = (_Float16)(n0 - (float)h0);
  lo[base + tid + 256] = (_Float16)(n1 - (float)h1);
}

// Per-lane A/B fragment load for v_wmma_f32_16x16x32_f16.
// Lane L: row = L%16, h = L/16; K pattern = {8h..8h+7} U {16+8h..16+8h+7}.
// rowBase points at the lane's matrix row; two 16B aligned loads.
__device__ __forceinline__ v16h esc_load_frag(const _Float16* __restrict__ rowBase,
                                              int kc, int h) {
  const _Float16* p = rowBase + kc * 32 + h * 8;
  const v8h x0 = *(const v8h*)(p);
  const v8h x1 = *(const v8h*)(p + 16);
  v16h r;
#pragma unroll
  for (int i = 0; i < 8; ++i) { r[i] = x0[i]; r[i + 8] = x1[i]; }
  return r;
}

// ---------------- kernel 2: WMMA Gram tiles + masked row sums -------------
// Grid: 512 blocks (one I-tile of 16 rows each), 8 waves/block (J-chunks).
__global__ void __launch_bounds__(256)
ESCL_wmma_kernel(const _Float16* __restrict__ hi, const _Float16* __restrict__ lo,
                 const int* __restrict__ labels,
                 float* __restrict__ posPart, float* __restrict__ totPart) {
  const int lane = threadIdx.x & 31;
  const int wv   = threadIdx.x >> 5;   // J-chunk id 0..7
  const int m = lane & 15;
  const int h = lane >> 4;
  const int iBase = blockIdx.x * 16;

  const _Float16* aHiRow = hi + (size_t)(iBase + m) * DIMS;
  const _Float16* aLoRow = lo + (size_t)(iBase + m) * DIMS;

  int labRow[8];
#pragma unroll
  for (int r = 0; r < 8; ++r) labRow[r] = labels[iBase + r + 8 * h];

  float posAcc[8], totAcc[8];
#pragma unroll
  for (int r = 0; r < 8; ++r) { posAcc[r] = 0.0f; totAcc[r] = 0.0f; }

  for (int js = 0; js < STRIPS_PER_CHUNK; ++js) {
    const int jBase = (wv * STRIPS_PER_CHUNK + js) * 64;

    const _Float16* bHiRow[4];
    const _Float16* bLoRow[4];
#pragma unroll
    for (int t = 0; t < 4; ++t) {
      bHiRow[t] = hi + (size_t)(jBase + t * 16 + m) * DIMS;
      bLoRow[t] = lo + (size_t)(jBase + t * 16 + m) * DIMS;
    }

    v8f acc[4] = {};
    for (int kc = 0; kc < KCHUNKS; ++kc) {
      const v16h aH = esc_load_frag(aHiRow, kc, h);
      const v16h aL = esc_load_frag(aLoRow, kc, h);
#pragma unroll
      for (int t = 0; t < 4; ++t) {
        const v16h bH = esc_load_frag(bHiRow[t], kc, h);
        const v16h bL = esc_load_frag(bLoRow[t], kc, h);
        acc[t] = __builtin_amdgcn_wmma_f32_16x16x32_f16(
            false, aH, false, bH, (short)0, acc[t], false, false);
        acc[t] = __builtin_amdgcn_wmma_f32_16x16x32_f16(
            false, aH, false, bL, (short)0, acc[t], false, false);
        acc[t] = __builtin_amdgcn_wmma_f32_16x16x32_f16(
            false, aL, false, bH, (short)0, acc[t], false, false);
      }
    }

    // epilogue: exp, diagonal mask, label mask (+eps), per-row accumulate
#pragma unroll
    for (int t = 0; t < 4; ++t) {
      const int nGlob  = jBase + t * 16 + m;
      const int labCol = labels[nGlob];
#pragma unroll
      for (int r = 0; r < 8; ++r) {
        const int mGlob = iBase + r + 8 * h;
        float sim = __expf(acc[t][r] * INV_T);
        if (mGlob == nGlob) sim = 0.0f;                     // logits_mask
        const float w = (labRow[r] == labCol ? 1.0f : 0.0f) + MASK_EPS;
        posAcc[r] = fmaf(sim, w, posAcc[r]);
        totAcc[r] += sim;
      }
    }
  }

  // reduce across the 16 lanes of each half-wave (row m=r+8h lives there)
#pragma unroll
  for (int r = 0; r < 8; ++r) {
    float p = posAcc[r], tv = totAcc[r];
#pragma unroll
    for (int sft = 1; sft < 16; sft <<= 1) {
      p  += __shfl_xor(p,  sft, 32);
      tv += __shfl_xor(tv, sft, 32);
    }
    if (m == 0) {
      const int row = iBase + r + 8 * h;
      posPart[(size_t)wv * NROWS + row] = p;   // no atomics: deterministic
      totPart[(size_t)wv * NROWS + row] = tv;
    }
  }
}

// ---------------- kernel 3: loss = -mean(log(pos/tot)) -------------------
__global__ void __launch_bounds__(256)
ESCL_finalize_kernel(const float* __restrict__ posPart,
                     const float* __restrict__ totPart,
                     float* __restrict__ out) {
  __shared__ float red[256];
  const int tid = threadIdx.x;
  float s = 0.0f;
  for (int i = tid; i < NROWS; i += 256) {
    float p = 0.0f, t = 0.0f;
#pragma unroll
    for (int c = 0; c < JCHUNKS; ++c) {
      p += posPart[(size_t)c * NROWS + i];
      t += totPart[(size_t)c * NROWS + i];
    }
    s += logf(p / t);
  }
  red[tid] = s;
  __syncthreads();
  for (int st = 128; st > 0; st >>= 1) {
    if (tid < st) red[tid] += red[tid + st];
    __syncthreads();
  }
  if (tid == 0) {
    const float loss = -red[0] / (float)NROWS;
    out[0] = isfinite(loss) ? loss : 0.0f;
  }
}

extern "C" void kernel_launch(void* const* d_in, const int* in_sizes, int n_in,
                              void* d_out, int out_size, void* d_ws, size_t ws_size,
                              hipStream_t stream) {
  (void)in_sizes; (void)n_in; (void)out_size; (void)ws_size;
  const float* feat   = (const float*)d_in[0];
  const int*   labels = (const int*)d_in[1];   // integer input per harness contract
  float* out = (float*)d_out;

  char* ws = (char*)d_ws;
  const size_t halfBytes = (size_t)NROWS * DIMS * sizeof(_Float16); // 8 MB
  _Float16* hi = (_Float16*)ws;
  _Float16* lo = (_Float16*)(ws + halfBytes);
  float* posPart = (float*)(ws + 2 * halfBytes);                    // 8*8192 f32
  float* totPart = posPart + (size_t)JCHUNKS * NROWS;               // 8*8192 f32

  ESCL_norm_kernel<<<NROWS, 256, 0, stream>>>(feat, hi, lo);
  ESCL_wmma_kernel<<<NROWS / 16, 256, 0, stream>>>(hi, lo, labels, posPart, totPart);
  ESCL_finalize_kernel<<<1, 256, 0, stream>>>(posPart, totPart, out);
}
